// PointTransformer_MultiFlowCondAdapter_62878321214037
// MI455X (gfx1250) — matrix operand order repaired and checked by reading, not compiled, not run on hardware
//
#include <hip/hip_runtime.h>
#include <math.h>

// ---------------- problem constants ----------------
constexpr int cB = 2, cN = 4096, cC = 256, cA = 64, cK = 16, cCH = 512;
constexpr int DEPTH = 2;

// ---------------- WMMA / vector types ----------------
typedef __attribute__((ext_vector_type(16))) __bf16 v16bf;
typedef __attribute__((ext_vector_type(8)))  __bf16 v8bf;
typedef __attribute__((ext_vector_type(4)))  __bf16 v4bf;
typedef __attribute__((ext_vector_type(8)))  float  v8f;
typedef __attribute__((ext_vector_type(4)))  float  f4v;
union bfrag16 { v16bf v; v8bf h[2]; };

// =====================================================================
// Generic GEMM: O[M,Nn] = epi(A[M,K] x W[K,Nn] + bias) (+ resid), bf16 WMMA
// Block tile 128x64, 8 waves of 32x32 (2x2 wmma 16x16x32).
// Double-buffered LDS, software-pipelined global->LDS staging.
// Nn is compile-time; M % 128 == 0, K % 32 == 0 at all call sites.
// =====================================================================
#define BM 128
#define BN 64
#define BK 32

template<int Nn, int EPI, bool HASR> // EPI: 0 none, 1 relu
__global__ __launch_bounds__(256)
void k_gemm(const float* __restrict__ Aall, const float* __restrict__ Wall,
            const float* __restrict__ Ball, const float* __restrict__ Rall,
            float* __restrict__ Oall,
            int M, int Kd,
            long bsA, long bsW, long bsB, long bsR, long bsO)
{
    const int bz = blockIdx.z;
    const float* A = Aall + bsA * bz;
    const float* W = Wall + bsW * bz;
    const float* bias = Ball + bsB * bz;
    const float* R = HASR ? (Rall + bsR * bz) : nullptr;
    float* O = Oall + bsO * bz;

    __shared__ __bf16 As[2][BM][BK];   // row-major   [m][k]
    __shared__ __bf16 Bs[2][BN][BK];   // transposed  [n][k]

    const int m0 = blockIdx.y * BM;
    const int n0 = blockIdx.x * BN;
    const int tid = threadIdx.x;
    const int wave = tid >> 5;
    const int lane = tid & 31;
    const int wm = wave >> 1;           // 0..3 (M dir)
    const int wn = wave & 1;            // 0..1 (N dir)
    const int half = lane >> 4;
    const int ml = lane & 15;

    f4v aR[4], bR[2];                   // staging registers (pipeline)

    auto loadTiles = [&](int k0) {
#pragma unroll
        for (int i = 0; i < 4; ++i) {
            int c4 = tid + i * 256;               // 0..1023 float4 chunks
            int r = c4 >> 3, cc = (c4 & 7) * 4;   // 8 chunks / 32-wide row
            aR[i] = *(const f4v*)(A + (long)(m0 + r) * Kd + k0 + cc);
        }
#pragma unroll
        for (int i = 0; i < 2; ++i) {
            int c4 = tid + i * 256;               // 0..511 float4 chunks
            int kk = c4 >> 4, nn = (c4 & 15) * 4;
            bR[i] = *(const f4v*)(W + (long)(k0 + kk) * Nn + n0 + nn);
        }
    };
    auto storeTiles = [&](int buf) {
#pragma unroll
        for (int i = 0; i < 4; ++i) {
            int c4 = tid + i * 256;
            int r = c4 >> 3, cc = (c4 & 7) * 4;
            v4bf b4;
            b4[0] = (__bf16)aR[i][0]; b4[1] = (__bf16)aR[i][1];
            b4[2] = (__bf16)aR[i][2]; b4[3] = (__bf16)aR[i][3];
            *(v4bf*)(&As[buf][r][cc]) = b4;       // 8-byte ds_store
        }
#pragma unroll
        for (int i = 0; i < 2; ++i) {
            int c4 = tid + i * 256;
            int kk = c4 >> 4, nn = (c4 & 15) * 4;
            Bs[buf][nn + 0][kk] = (__bf16)bR[i][0];
            Bs[buf][nn + 1][kk] = (__bf16)bR[i][1];
            Bs[buf][nn + 2][kk] = (__bf16)bR[i][2];
            Bs[buf][nn + 3][kk] = (__bf16)bR[i][3];
        }
    };

    v8f acc[2][2] = {};

    loadTiles(0);
    storeTiles(0);
    __syncthreads();

    const int nt = Kd / BK;
    for (int kt = 0; kt < nt; ++kt) {
        const bool hasNext = (kt + 1 < nt);
        if (hasNext) loadTiles((kt + 1) * BK);    // overlap with WMMA below

        const __bf16 (*Ab)[BK] = As[kt & 1];
        const __bf16 (*Bb)[BK] = Bs[kt & 1];
        bfrag16 af[2], bf[2];
#pragma unroll
        for (int t = 0; t < 2; ++t) {
            const __bf16* ap = &Ab[wm * 32 + t * 16 + ml][half * 8];
            af[t].h[0] = *(const v8bf*)ap;          // K 0..7  (or 8..15)
            af[t].h[1] = *(const v8bf*)(ap + 16);   // K 16..23 (or 24..31)
            const __bf16* bp = &Bb[wn * 32 + t * 16 + ml][half * 8];
            bf[t].h[0] = *(const v8bf*)bp;          // K 0..7  (or 8..15)
            bf[t].h[1] = *(const v8bf*)(bp + 16);   // K 16..23 (or 24..31)
        }
#pragma unroll
        for (int i = 0; i < 2; ++i)
#pragma unroll
            for (int j = 0; j < 2; ++j)
                acc[i][j] = __builtin_amdgcn_wmma_f32_16x16x32_bf16(
                    false, af[i].v, false, bf[j].v, (short)0, acc[i][j], false, false);

        if (hasNext) storeTiles((kt + 1) & 1);
        __syncthreads();
    }

    // ---- epilogue + store: acc element r -> (M = r + 8*half, N = ml) ----
#pragma unroll
    for (int i = 0; i < 2; ++i) {
#pragma unroll
        for (int j = 0; j < 2; ++j) {
            int ncol = n0 + wn * 32 + j * 16 + ml;
            float bv = bias[ncol];
            long obase = (long)(m0 + wm * 32 + i * 16 + half * 8) * Nn + ncol;
            float* op = O + obase;
            const float* rp = HASR ? (R + obase) : nullptr;
#pragma unroll
            for (int r = 0; r < 8; ++r) {
                float v = acc[i][j][r] + bv;
                if (EPI == 1) v = fmaxf(v, 0.0f);
                if (HASR) v += rp[r * Nn];
                op[r * Nn] = v;
            }
        }
    }
}

// =====================================================================
// LayerNorm (wave per row) with fused FiLM / GELU / residual epilogues
// MODE bits: 1 = FiLM, 2 = exact GELU, 4 = residual add
// =====================================================================
template<int D, int MODE>
__global__ __launch_bounds__(256)
void k_lnorm(const float* __restrict__ X, float* __restrict__ Y,
             const float* __restrict__ g, const float* __restrict__ be,
             long gbStride,
             const float* __restrict__ film, int filmHalf,
             const float* __restrict__ resid,
             int Mrows, int rowsPerBatch)
{
    constexpr int CNT = D / 32;
    int row = blockIdx.x * 8 + (threadIdx.x >> 5);
    if (row >= Mrows) return;
    int lane = threadIdx.x & 31;
    int batch = row / rowsPerBatch;
    const float* x = X + (long)row * D;

    float v[CNT];
    float s = 0.0f;
#pragma unroll
    for (int i = 0; i < CNT; ++i) { v[i] = x[lane + i * 32]; s += v[i]; }
#pragma unroll
    for (int o = 16; o > 0; o >>= 1) s += __shfl_xor(s, o, 32);
    float mean = s * (1.0f / D);
    float var = 0.0f;
#pragma unroll
    for (int i = 0; i < CNT; ++i) { float d = v[i] - mean; var += d * d; }
#pragma unroll
    for (int o = 16; o > 0; o >>= 1) var += __shfl_xor(var, o, 32);
    float rstd = rsqrtf(var * (1.0f / D) + 1e-5f);

    const float* gg = g + gbStride * batch;
    const float* bb = be + gbStride * batch;
#pragma unroll
    for (int i = 0; i < CNT; ++i) {
        int c = lane + i * 32;
        float y = (v[i] - mean) * rstd * gg[c] + bb[c];
        if (MODE & 1) {
            float sh = film[batch * 2 * filmHalf + c];
            float sc = film[batch * 2 * filmHalf + filmHalf + c];
            y = y * (1.0f + sc) + sh;
        }
        if (MODE & 2) y = 0.5f * y * (1.0f + erff(y * 0.70710678118f));
        if (MODE & 4) y += resid[(long)row * D + c];
        Y[(long)row * D + c] = y;
    }
}

// =====================================================================
// K=3 linear (proj_in / pos1) -- thread per output element
// =====================================================================
template<int EPI>
__global__ __launch_bounds__(256)
void k_lin3(const float* __restrict__ X, const float* __restrict__ W,
            const float* __restrict__ b, float* __restrict__ O, long M, int Nn)
{
    long i = (long)blockIdx.x * 256 + threadIdx.x;
    if (i >= M * Nn) return;
    long m = i / Nn; int n = (int)(i - m * Nn);
    const float* x = X + m * 3;
    float v = b[n] + x[0] * W[n] + x[1] * W[Nn + n] + x[2] * W[2 * Nn + n];
    if (EPI == 1) v = fmaxf(v, 0.0f);
    O[i] = v;
}

// proj_out (N==1): wave-per-row dot product
__global__ __launch_bounds__(256)
void k_dot(const float* __restrict__ X, const float* __restrict__ W,
           const float* __restrict__ b, float* __restrict__ O, int M, int D)
{
    int row = blockIdx.x * 8 + (threadIdx.x >> 5);
    if (row >= M) return;
    int lane = threadIdx.x & 31;
    float s = 0.0f;
    for (int c = lane; c < D; c += 32) s += X[(long)row * D + c] * W[c];
#pragma unroll
    for (int o = 16; o > 0; o >>= 1) s += __shfl_xor(s, o, 32);
    if (lane == 0) O[row] = s + b[0];
}

// =====================================================================
// Brute-force KNN, LDS tiled, branchless top-16 insertion
// =====================================================================
__global__ __launch_bounds__(256)
void k_knn(const float* __restrict__ pos, int* __restrict__ idx)
{
    const int blocksPerBatch = cN / 256;
    int b = blockIdx.x / blocksPerBatch;
    int q = (blockIdx.x % blocksPerBatch) * 256 + threadIdx.x;
    const float* pb = pos + (long)b * cN * 3;
    float qx = pb[q * 3 + 0], qy = pb[q * 3 + 1], qz = pb[q * 3 + 2];

    float bd[cK]; int bi[cK];
#pragma unroll
    for (int k = 0; k < cK; ++k) { bd[k] = 3.4e38f; bi[k] = 0; }

    __shared__ float sx[256], sy[256], sz[256];
    for (int tile = 0; tile < cN; tile += 256) {
        __syncthreads();
        int t = threadIdx.x;
        sx[t] = pb[(tile + t) * 3 + 0];
        sy[t] = pb[(tile + t) * 3 + 1];
        sz[t] = pb[(tile + t) * 3 + 2];
        __syncthreads();
        for (int j = 0; j < 256; ++j) {
            float dx = qx - sx[j], dy = qy - sy[j], dz = qz - sz[j];
            float d = dx * dx + dy * dy + dz * dz;
            if (d < bd[cK - 1]) {
                float cd = d; int ci = tile + j;
#pragma unroll
                for (int s = 0; s < cK; ++s) {
                    bool less = cd < bd[s];
                    float td = less ? bd[s] : cd;
                    int   ti = less ? bi[s] : ci;
                    bd[s] = less ? cd : bd[s];
                    bi[s] = less ? ci : bi[s];
                    cd = td; ci = ti;
                }
            }
        }
    }
#pragma unroll
    for (int k = 0; k < cK; ++k) idx[((long)b * cN + q) * cK + k] = bi[k];
}

// =====================================================================
// Adapter helpers
// =====================================================================
struct CondW {
    const float *W1[3], *b1[3], *g[3], *be[3], *W2[3], *b2[3];
};

// shift/scale [B, 2A] from cond MLP (route-selected branch), one block/batch
__global__ __launch_bounds__(128)
void k_cond(const float* __restrict__ cond, const int* __restrict__ route,
            CondW w, float* __restrict__ ss)
{
    int b = blockIdx.x, t = threadIdx.x;
    int r = route[b]; r = (r < 0) ? 0 : (r > 2 ? 2 : r);
    __shared__ float h[128];
    float v = w.b1[r][t] + cond[b * 2 + 0] * w.W1[r][t] + cond[b * 2 + 1] * w.W1[r][128 + t];
    h[t] = v;
    __syncthreads();
    float m = 0.0f;
    for (int i = 0; i < 128; ++i) m += h[i];
    m *= (1.0f / 128.0f);
    float var = 0.0f;
    for (int i = 0; i < 128; ++i) { float d = h[i] - m; var += d * d; }
    var *= (1.0f / 128.0f);
    float hn = (v - m) * rsqrtf(var + 1e-5f) * w.g[r][t] + w.be[r][t];
    __syncthreads();
    h[t] = hn;
    __syncthreads();
    float o = w.b2[r][t];
    for (int i = 0; i < 128; ++i) o += h[i] * w.W2[r][(long)i * 128 + t];
    ss[b * 128 + t] = o;
}

// copy route-selected weight tensor per batch
__global__ __launch_bounds__(256)
void k_pick(const float* __restrict__ s0, const float* __restrict__ s1,
            const float* __restrict__ s2, const int* __restrict__ route,
            float* __restrict__ dst, int n)
{
    int b = blockIdx.y;
    int r = route[b]; r = (r < 0) ? 0 : (r > 2 ? 2 : r);
    const float* s = (r == 0) ? s0 : ((r == 1) ? s1 : s2);
    for (int i = blockIdx.x * 256 + threadIdx.x; i < n; i += gridDim.x * 256)
        dst[(long)b * n + i] = s[i];
}

// =====================================================================
// Attention pieces (chunked over N by cCH points)
// =====================================================================
__global__ __launch_bounds__(256)
void k_pd(const float* __restrict__ pos, const int* __restrict__ idx,
          float* __restrict__ pd, int n0)
{
    int i = blockIdx.x * 256 + threadIdx.x;        // over B*CH*K
    if (i >= cB * cCH * cK) return;
    int k = i & (cK - 1); int rest = i >> 4;
    int ci = rest % cCH;  int b = rest / cCH;
    int n = n0 + ci;
    int j = idx[((long)b * cN + n) * cK + k];
    const float* pb = pos + (long)b * cN * 3;
    pd[(long)i * 3 + 0] = pb[n * 3 + 0] - pb[j * 3 + 0];
    pd[(long)i * 3 + 1] = pb[n * 3 + 1] - pb[j * 3 + 1];
    pd[(long)i * 3 + 2] = pb[n * 3 + 2] - pb[j * 3 + 2];
}

// rel = k_gathered - q + pe   (kv rows have stride 2C; k part = first C)
__global__ __launch_bounds__(256)
void k_rel(const float* __restrict__ kv, const float* __restrict__ q,
           const float* __restrict__ pe, const int* __restrict__ idx,
           float* __restrict__ rel, int n0)
{
    long i = (long)blockIdx.x * 256 + threadIdx.x; // over B*CH*K*C
    if (i >= (long)cB * cCH * cK * cC) return;
    int c = (int)(i & (cC - 1)); long r = i >> 8;
    int k = (int)(r & (cK - 1)); r >>= 4;
    int ci = (int)(r % cCH); int b = (int)(r / cCH);
    int n = n0 + ci;
    int j = idx[((long)b * cN + n) * cK + k];
    float kval = kv[((long)b * cN + j) * (2 * cC) + c];
    rel[i] = kval - q[((long)b * cN + n) * cC + c] + pe[i];
}

// softmax over K (per b,n,c) + aggregate sum_k a*(v_gathered + pe)
__global__ __launch_bounds__(256)
void k_smax(const float* __restrict__ a, const float* __restrict__ pe,
            const float* __restrict__ kv, const int* __restrict__ idx,
            float* __restrict__ agg, int n0)
{
    long i = (long)blockIdx.x * 256 + threadIdx.x; // over B*CH*C
    if (i >= (long)cB * cCH * cC) return;
    int c = (int)(i & (cC - 1)); long r = i >> 8;
    int ci = (int)(r % cCH); int b = (int)(r / cCH);
    int n = n0 + ci;
    long base = ((long)(b * cCH + ci) * cK) * cC + c;

    float av[cK]; float mx = -3.4e38f;
#pragma unroll
    for (int k = 0; k < cK; ++k) { av[k] = a[base + (long)k * cC]; mx = fmaxf(mx, av[k]); }
    float s = 0.0f;
#pragma unroll
    for (int k = 0; k < cK; ++k) { av[k] = expf(av[k] - mx); s += av[k]; }
    float inv = 1.0f / s;
    float acc = 0.0f;
#pragma unroll
    for (int k = 0; k < cK; ++k) {
        int j = idx[((long)b * cN + n) * cK + k];
        float v = kv[((long)b * cN + j) * (2 * cC) + cC + c];
        acc += av[k] * inv * (v + pe[base + (long)k * cC]);
    }
    agg[((long)b * cN + n) * cC + c] = acc;
}

// =====================================================================
// Host orchestration
// =====================================================================
// Input layout (insertion order flatten):
// 0 pos, 1 cond, 2 route,
// 3 proj_in.W, 4 proj_in.b, 5 proj_out.W, 6 proj_out.b,
// layer L at base 7+64*L:
//   adapter a at +14*a: 0 inW 1 inb 2 inlng 3 inlnb 4 outW 5 outb 6 outlng
//                       7 outlnb 8 c1W 9 c1b 10 clng 11 clnb 12 c2W 13 c2b
//   layer-level at +42: 0 fcqW 1 fcqb 2 fckvW 3 fckvb 4 pos1W 5 pos1b
//                       6 pos2W 7 pos2b 8 attn1W 9 attn1b 10 attn2W 11 attn2b
//                       12 fcoW 13 fcob 14 ffn1W 15 ffn1b 16 ln1g 17 ln1b
//                       18 ffn2W 19 ffn2b 20 ln2g 21 ln2b

template<int Nn, int EPI, bool HASR>
static inline void launch_gemm(const float* A, const float* W, const float* bias,
                               const float* R, float* O, int M, int Kd,
                               long bsA, long bsW, long bsB, long bsR, long bsO,
                               int nb, hipStream_t stream)
{
    dim3 g(Nn / BN, M / BM, nb), blk(256);
    k_gemm<Nn, EPI, HASR><<<g, blk, 0, stream>>>(A, W, bias, R, O, M, Kd,
                                                 bsA, bsW, bsB, bsR, bsO);
}

extern "C" void kernel_launch(void* const* d_in, const int* in_sizes, int n_in,
                              void* d_out, int out_size, void* d_ws, size_t ws_size,
                              hipStream_t stream)
{
    auto FP = [&](int i) { return (const float*)d_in[i]; };
    const float* pos  = FP(0);
    const float* cond = FP(1);
    const int*   route = (const int*)d_in[2];

    // ---- workspace carve-up (floats) ----
    float* ws = (float*)d_ws;
    size_t off = 0;
    auto alloc = [&](size_t n) { float* p = ws + off; off += n; return p; };
    float* xbuf = alloc((size_t)cB * cN * cC);          // current features
    float* ybuf = alloc((size_t)cB * cN * 4 * cC);      // kv / ffn hidden / adapter tmp
    float* zbuf = alloc((size_t)cB * cN * cC);          // adapter z / attention agg
    float* qbuf = alloc((size_t)cB * cN * cC);
    float* pe   = alloc((size_t)cB * cCH * cK * cC);
    float* t1   = alloc((size_t)cB * cCH * cK * cC);
    float* t2   = alloc((size_t)cB * cCH * cK * cC);
    float* pd   = alloc((size_t)cB * cCH * cK * 3);
    float* ss   = alloc((size_t)cB * 2 * cA);
    float* wInW  = alloc((size_t)cB * cC * cA);
    float* wInB  = alloc((size_t)cB * cA);
    float* wInG  = alloc((size_t)cB * cA);
    float* wInBe = alloc((size_t)cB * cA);
    float* wOutW = alloc((size_t)cB * cA * cC);
    float* wOutB = alloc((size_t)cB * cC);
    float* wOutG = alloc((size_t)cB * cC);
    float* wOutBe= alloc((size_t)cB * cC);
    int*   idxb  = (int*)alloc((size_t)cB * cN * cK);

    const int MN = cB * cN;

    // ---- proj_in + knn ----
    k_lin3<0><<<dim3(((long)MN * cC) / 256), 256, 0, stream>>>(pos, FP(3), FP(4), xbuf, MN, cC);
    k_knn<<<dim3(MN / 256), 256, 0, stream>>>(pos, idxb);

    for (int L = 0; L < DEPTH; ++L) {
        const int lb = 7 + 64 * L;
        const int la = lb + 42;

        // ---------- router / adapter (route-selected branch only) ----------
        CondW cw;
        for (int a = 0; a < 3; ++a) {
            int ab = lb + a * 14;
            cw.W1[a] = FP(ab + 8);  cw.b1[a] = FP(ab + 9);
            cw.g[a]  = FP(ab + 10); cw.be[a] = FP(ab + 11);
            cw.W2[a] = FP(ab + 12); cw.b2[a] = FP(ab + 13);
        }
        k_cond<<<dim3(cB), dim3(128), 0, stream>>>(cond, route, cw, ss);

        auto pick = [&](int o, float* dst, int n) {
            k_pick<<<dim3((n + 255) / 256, cB), 256, 0, stream>>>(
                FP(lb + o), FP(lb + 14 + o), FP(lb + 28 + o), route, dst, n);
        };
        pick(0, wInW, cC * cA);  pick(1, wInB, cA);
        pick(2, wInG, cA);       pick(3, wInBe, cA);
        pick(4, wOutW, cA * cC); pick(5, wOutB, cC);
        pick(6, wOutG, cC);      pick(7, wOutBe, cC);

        // z = LN(in_proj(x)) ; FiLM ; z = LN(out_proj(z)) ; x += z
        launch_gemm<cA, 0, false>(xbuf, wInW, wInB, nullptr, zbuf, cN, cC,
            (long)cN * cC, (long)cC * cA, cA, 0, (long)cN * cA, cB, stream);
        k_lnorm<cA, 1><<<dim3(MN / 8), 256, 0, stream>>>(
            zbuf, zbuf, wInG, wInBe, (long)cA, ss, cA, nullptr, MN, cN);
        launch_gemm<cC, 0, false>(zbuf, wOutW, wOutB, nullptr, ybuf, cN, cA,
            (long)cN * cA, (long)cA * cC, cC, 0, (long)cN * cC, cB, stream);
        k_lnorm<cC, 4><<<dim3(MN / 8), 256, 0, stream>>>(
            ybuf, xbuf, wOutG, wOutBe, (long)cC, nullptr, 0, xbuf, MN, cN);

        // ---------- neighbor attention ----------
        launch_gemm<cC, 0, false>(xbuf, FP(la + 0), FP(la + 1), nullptr, qbuf,
                                  MN, cC, 0, 0, 0, 0, 0, 1, stream);
        launch_gemm<2 * cC, 0, false>(xbuf, FP(la + 2), FP(la + 3), nullptr, ybuf,
                                      MN, cC, 0, 0, 0, 0, 0, 1, stream);

        const int Mc = cB * cCH * cK; // 16384 rows per chunk
        for (int n0 = 0; n0 < cN; n0 += cCH) {
            k_pd<<<dim3((cB * cCH * cK) / 256), 256, 0, stream>>>(pos, idxb, pd, n0);
            k_lin3<1><<<dim3(((long)Mc * cC) / 256), 256, 0, stream>>>(
                pd, FP(la + 4), FP(la + 5), t1, Mc, cC);                    // relu(pos1(pd))
            launch_gemm<cC, 0, false>(t1, FP(la + 6), FP(la + 7), nullptr, pe,
                                      Mc, cC, 0, 0, 0, 0, 0, 1, stream);    // pe = pos2(.)
            k_rel<<<dim3(((long)Mc * cC) / 256), 256, 0, stream>>>(
                ybuf, qbuf, pe, idxb, t1, n0);                              // rel
            launch_gemm<cC, 1, false>(t1, FP(la + 8), FP(la + 9), nullptr, t2,
                                      Mc, cC, 0, 0, 0, 0, 0, 1, stream);    // relu(attn1)
            launch_gemm<cC, 0, false>(t2, FP(la + 10), FP(la + 11), nullptr, t1,
                                      Mc, cC, 0, 0, 0, 0, 0, 1, stream);    // attn2
            k_smax<<<dim3((cB * cCH * cC) / 256), 256, 0, stream>>>(
                t1, pe, ybuf, idxb, zbuf, n0);                              // softmax+agg
        }
        launch_gemm<cC, 0, true>(zbuf, FP(la + 12), FP(la + 13), xbuf, xbuf,
                                 MN, cC, 0, 0, 0, 0, 0, 1, stream);         // x += fc_out

        // ---------- FFN ----------
        launch_gemm<4 * cC, 0, false>(xbuf, FP(la + 14), FP(la + 15), nullptr, ybuf,
                                      MN, cC, 0, 0, 0, 0, 0, 1, stream);
        k_lnorm<4 * cC, 2><<<dim3(MN / 8), 256, 0, stream>>>(
            ybuf, ybuf, FP(la + 16), FP(la + 17), 0, nullptr, 0, nullptr, MN, cN);
        launch_gemm<cC, 0, false>(ybuf, FP(la + 18), FP(la + 19), nullptr, zbuf,
                                  MN, 4 * cC, 0, 0, 0, 0, 0, 1, stream);
        k_lnorm<cC, 4><<<dim3(MN / 8), 256, 0, stream>>>(
            zbuf, xbuf, FP(la + 20), FP(la + 21), 0, nullptr, 0, xbuf, MN, cN);
    }

    // ---- proj_out ----
    k_dot<<<dim3(MN / 8), 256, 0, stream>>>(xbuf, FP(5), FP(6), (float*)d_out, MN, cC);

    (void)in_sizes; (void)n_in; (void)out_size; (void)ws_size;
}
